// SSIM_10041633538282
// MI455X (gfx1250) — matrix operand order repaired and checked by reading, not compiled, not run on hardware
//
#include <hip/hip_runtime.h>
#include <hip/hip_bf16.h>

// ---------------------------------------------------------------------------
// Types & helpers
// ---------------------------------------------------------------------------
typedef unsigned short ushort_t;
typedef __attribute__((ext_vector_type(16))) __bf16 v16bf;
typedef __attribute__((ext_vector_type(8)))  float  v8f;

struct Q32B { uint4 a, b; };

static __device__ __forceinline__ v16bf make_frag(uint4 a, uint4 b) {
    Q32B q{a, b};
    return __builtin_bit_cast(v16bf, q);
}

// round-to-nearest-even f32 -> bf16 (bit manipulation, no __bf16 arithmetic)
static __device__ __forceinline__ ushort_t f2bf(float f) {
    unsigned u = __float_as_uint(f);
    unsigned r = u + 0x7FFFu + ((u >> 16) & 1u);
    return (ushort_t)(r >> 16);
}
static __device__ __forceinline__ float bf2f(ushort_t h) {
    return __uint_as_float(((unsigned)h) << 16);
}
static __device__ __forceinline__ float sigmoidf_(float x) {
    return 1.0f / (1.0f + __expf(-x));
}

#define WMMA_BF16(A, B, C) \
    __builtin_amdgcn_wmma_f32_16x16x32_bf16(false, (A), false, (B), (short)0, (C), false, false)

// Problem dims
#define NB   64     // batch
#define NS   512    // seq len
#define NI   64     // input size
#define NE   256    // enc hidden
#define ND   256    // dec hidden
#define KENC 320    // NI + NE
#define KDEC 544    // 513 padded to 17*32

// ---------------------------------------------------------------------------
// Prep kernels: bf16 packing / constant precompute
// ---------------------------------------------------------------------------
__global__ void k_cvt_x(const float* __restrict__ x, ushort_t* __restrict__ xb) {
    int i = blockIdx.x * 256 + threadIdx.x;
    if (i < NB * NS * NI) xb[i] = f2bf(x[i]);
}

// Wcomb[N=1024][K=320]: cols 0..63 = Wih[n][k], cols 64..319 = Whh[n][k-64]
__global__ void k_pack_w(const float* __restrict__ Wih, const float* __restrict__ Whh,
                         ushort_t* __restrict__ Wo) {
    int i = blockIdx.x * 256 + threadIdx.x;
    if (i >= 1024 * KENC) return;
    int n = i / KENC, k = i % KENC;
    float v = (k < NI) ? Wih[n * NI + k] : Whh[n * NE + (k - NI)];
    Wo[i] = f2bf(v);
}

// W_attn_enc[256][512] bf16 from W_attn[:, 256:768]
__global__ void k_cvt_wattn(const float* __restrict__ W_attn, ushort_t* __restrict__ Wo) {
    int i = blockIdx.x * 256 + threadIdx.x;
    if (i >= 256 * 512) return;
    int d = i / 512, e = i % 512;
    Wo[i] = f2bf(W_attn[d * 768 + 256 + e]);
}

// Wd[1024][544] bf16 from Wih_d[1024][513], zero-padded
__global__ void k_cvt_wd(const float* __restrict__ Wih_d, ushort_t* __restrict__ Wo) {
    int i = blockIdx.x * 256 + threadIdx.x;
    if (i >= 1024 * KDEC) return;
    int n = i / KDEC, k = i % KDEC;
    Wo[i] = (k < 513) ? f2bf(Wih_d[n * 513 + k]) : (ushort_t)0;
}

// decoder state init: h_attn = dec_h0, y_buf = 0, dec_in = 0 (incl. K padding)
__global__ void k_init_dec(const float* __restrict__ dec_h0, float* __restrict__ h_attn,
                           float* __restrict__ y_buf, ushort_t* __restrict__ dec_in) {
    int i = blockIdx.x * 256 + threadIdx.x;
    if (i < NB * ND) h_attn[i] = dec_h0[i];
    if (i < NB) y_buf[i] = 0.0f;
    if (i < NB * KDEC) dec_in[i] = 0;
}

// zc[64][1024] = b_d + dec_h0 @ Whh_d.T  (constant across decoder steps)
__global__ void k_zc(const float* __restrict__ dec_h0, const float* __restrict__ Whh_d,
                     const float* __restrict__ b_d, float* __restrict__ zc) {
    int i = blockIdx.x * 256 + threadIdx.x;  // 64*1024 total
    int b = i >> 10, n = i & 1023;
    float s = b_d[n];
    const float* h = dec_h0 + b * ND;
    const float* w = Whh_d + n * ND;
#pragma unroll 8
    for (int e = 0; e < ND; ++e) s += h[e] * w[e];
    zc[i] = s;
}

// ---------------------------------------------------------------------------
// Bidirectional encoder LSTM: one WG per direction, h in LDS (bf16), c in regs.
// z[64,1024] = [x_t, h] @ Wcomb.T via WMMA bf16 (K=320, 10 k-steps).
// Weights are deliberately re-read from L2 each step (opaque loop-variant base
// + unroll-1 K loops) so the compiler cannot hoist 2.5KB/lane of fragments
// into scratch spills.
// ---------------------------------------------------------------------------
__global__ __launch_bounds__(1024) void enc_lstm(
    const ushort_t* __restrict__ x_bf,   // [64][512][64]
    const ushort_t* __restrict__ Wf,     // [1024][320]
    const ushort_t* __restrict__ Wb,
    const float* __restrict__ b_f, const float* __restrict__ b_b,
    const float* __restrict__ enc_h0,    // [2][64][256]
    const float* __restrict__ enc_c0,
    float* __restrict__ enc_out,         // [64][512][512] f32
    ushort_t* __restrict__ enc_out_bf)   // same layout, bf16
{
    const int dir = blockIdx.x;
    const ushort_t* W = dir ? Wb : Wf;
    const float* bias = dir ? b_b : b_f;

    __shared__ ushort_t h_sh[64][264];   // padded for bank spread; rows 16B aligned

    const int tid = threadIdx.x;
    const int wave = tid >> 5, lane = tid & 31;
    const int mtile = wave >> 3, gpair = wave & 7;
    const int m0 = mtile * 16;
    const int lrow = lane & 15, lhalf = lane >> 4;
    const int col0[2] = { gpair * 32, gpair * 32 + 16 };

    // init h (LDS, bf16)
    for (int e = tid; e < 64 * 256; e += 1024)
        h_sh[e >> 8][e & 255] = f2bf(enc_h0[dir * 16384 + e]);

    // init c (registers, tile-owned)
    v8f creg[2];
#pragma unroll
    for (int j = 0; j < 2; ++j)
#pragma unroll
        for (int r = 0; r < 8; ++r) {
            int row = m0 + lhalf * 8 + r;
            creg[j][r] = enc_c0[dir * 16384 + row * 256 + col0[j] + lrow];
        }
    // gate biases for this lane
    float bg[2][4];
#pragma unroll
    for (int j = 0; j < 2; ++j)
#pragma unroll
        for (int g = 0; g < 4; ++g)
            bg[j][g] = bias[g * 256 + col0[j] + lrow];

    // opaque zero: compiler cannot prove weight addresses are t-invariant,
    // so weight fragments are re-fetched from L2 instead of spilled to scratch
    int opq0;
    asm volatile("s_mov_b32 %0, 0" : "=s"(opq0));

    __syncthreads();

    for (int t = 0; t < NS; ++t) {
        const int tx = dir ? (NS - 1 - t) : t;
        const ushort_t* Wt = W + (size_t)(opq0 * t);   // == W, but loop-variant
        v8f acc[2][4] = {};

        // K phase 1: A columns 0..63 = x_t (global)
#pragma unroll 1
        for (int ks = 0; ks < 2; ++ks) {
            int kb = ks * 32 + lhalf * 16;
            const ushort_t* ap = &x_bf[((m0 + lrow) * NS + tx) * NI];
            uint4 a0 = *(const uint4*)(ap + kb);
            uint4 a1 = *(const uint4*)(ap + kb + 8);
            v16bf afrag = make_frag(a0, a1);
#pragma unroll
            for (int j = 0; j < 2; ++j)
#pragma unroll
                for (int g = 0; g < 4; ++g) {
                    const ushort_t* bp =
                        &Wt[(g * 256 + col0[j] + lrow) * KENC + ks * 32 + lhalf * 16];
                    uint4 b0 = *(const uint4*)(bp);
                    uint4 b1 = *(const uint4*)(bp + 8);
                    acc[j][g] = WMMA_BF16(afrag, make_frag(b0, b1), acc[j][g]);
                }
        }
        // K phase 2: A columns 64..319 = h (LDS)
#pragma unroll 1
        for (int ks = 0; ks < 8; ++ks) {
            int kb = ks * 32 + lhalf * 16;
            const ushort_t* ap = &h_sh[m0 + lrow][0];
            uint4 a0 = *(const uint4*)(ap + kb);
            uint4 a1 = *(const uint4*)(ap + kb + 8);
            v16bf afrag = make_frag(a0, a1);
#pragma unroll
            for (int j = 0; j < 2; ++j)
#pragma unroll
                for (int g = 0; g < 4; ++g) {
                    const ushort_t* bp =
                        &Wt[(g * 256 + col0[j] + lrow) * KENC + (ks + 2) * 32 + lhalf * 16];
                    uint4 b0 = *(const uint4*)(bp);
                    uint4 b1 = *(const uint4*)(bp + 8);
                    acc[j][g] = WMMA_BF16(afrag, make_frag(b0, b1), acc[j][g]);
                }
        }
        __syncthreads();   // everyone done reading h_sh

#pragma unroll
        for (int j = 0; j < 2; ++j)
#pragma unroll
            for (int r = 0; r < 8; ++r) {
                int row = m0 + lhalf * 8 + r;
                int col = col0[j] + lrow;
                float iv = sigmoidf_(acc[j][0][r] + bg[j][0]);
                float fv = sigmoidf_(acc[j][1][r] + bg[j][1]);
                float gv = tanhf   (acc[j][2][r] + bg[j][2]);
                float ov = sigmoidf_(acc[j][3][r] + bg[j][3]);
                float c = fv * creg[j][r] + iv * gv;
                float h = ov * tanhf(c);
                creg[j][r] = c;
                h_sh[row][col] = f2bf(h);
                int oidx = (row * NS + tx) * 512 + dir * 256 + col;
                enc_out[oidx] = h;
                enc_out_bf[oidx] = f2bf(h);
            }
        __syncthreads();   // h_sh updated for next step
    }
}

// ---------------------------------------------------------------------------
// enc_proj[M=32768][256] = enc_out_bf[M][512] @ Wattn_enc[256][512].T + b_attn
// one 16x16 tile per wave, K-loop of 16 WMMAs
// ---------------------------------------------------------------------------
__global__ __launch_bounds__(256) void proj_gemm(
    const ushort_t* __restrict__ A,   // [32768][512] bf16
    const ushort_t* __restrict__ Bw,  // [256][512]   bf16
    const float* __restrict__ b_attn,
    float* __restrict__ P)            // [32768][256] f32
{
    const int wave = threadIdx.x >> 5, lane = threadIdx.x & 31;
    const int lin = blockIdx.x * 8 + wave;         // 0..32767
    const int mt = lin >> 4, nt = lin & 15;
    const int lrow = lane & 15, lhalf = lane >> 4;

    v8f acc = {};
    const ushort_t* ap = &A[(mt * 16 + lrow) * 512];
    const ushort_t* bp = &Bw[(nt * 16 + lrow) * 512];
#pragma unroll 4
    for (int ks = 0; ks < 16; ++ks) {
        uint4 a0 = *(const uint4*)(ap + ks * 32 + lhalf * 16);
        uint4 a1 = *(const uint4*)(ap + ks * 32 + lhalf * 16 + 8);
        uint4 b0 = *(const uint4*)(bp + ks * 32 + lhalf * 16);
        uint4 b1 = *(const uint4*)(bp + ks * 32 + lhalf * 16 + 8);
        acc = WMMA_BF16(make_frag(a0, a1), make_frag(b0, b1), acc);
    }
    const int n = nt * 16 + lrow;
    const float bn = b_attn[n];
#pragma unroll
    for (int r = 0; r < 8; ++r) {
        int row = mt * 16 + lhalf * 8 + r;
        P[row * 256 + n] = acc[r] + bn;
    }
}

// ---------------------------------------------------------------------------
// Decoder attention step: one WG per batch element.
//  scores -> softmax -> context; builds bf16 dec_in = [y_prev, context, 0-pad]
// ---------------------------------------------------------------------------
__global__ __launch_bounds__(256) void attn_step(
    const float* __restrict__ enc_proj,  // [64][512][256] (bias folded)
    const float* __restrict__ enc_out,   // [64][512][512] f32
    const float* __restrict__ W_attn,    // [256][768] f32 (h part: cols 0..255)
    const float* __restrict__ vvec,      // [256]
    const float* __restrict__ h_attn,    // [64][256]
    const float* __restrict__ y_buf,     // [64]
    float* __restrict__ ctx,             // [64][512]
    ushort_t* __restrict__ dec_in)       // [64][544]
{
    const int b = blockIdx.x;
    const int tid = threadIdx.x;
    __shared__ float hbuf[256];
    __shared__ float hp[256];
    __shared__ float sc[512];
    __shared__ float red[256];

    hbuf[tid] = h_attn[b * ND + tid];
    __syncthreads();

    // h projection: hp[d] = sum_e h[e] * W_attn[d][e]
    {
        float s = 0.0f;
        const float* w = W_attn + tid * 768;
#pragma unroll 8
        for (int e = 0; e < 256; ++e) s += hbuf[e] * w[e];
        hp[tid] = s;
    }
    __syncthreads();

    // scores: sc[t] = sum_d v[d] * tanh(enc_proj[b,t,d] + hp[d])
    for (int t = tid; t < NS; t += 256) {
        const float* ep = enc_proj + ((size_t)(b * NS + t)) * 256;
        __builtin_prefetch(ep + 256 * 256, 0, 0);   // next t for this thread (L2)
        float s = 0.0f;
#pragma unroll 8
        for (int d = 0; d < 256; ++d) s += vvec[d] * tanhf(ep[d] + hp[d]);
        sc[t] = s;
    }
    __syncthreads();

    // softmax over t
    float m = -3.4e38f;
    for (int t = tid; t < NS; t += 256) m = fmaxf(m, sc[t]);
    red[tid] = m; __syncthreads();
    for (int s = 128; s > 0; s >>= 1) {
        if (tid < s) red[tid] = fmaxf(red[tid], red[tid + s]);
        __syncthreads();
    }
    const float mx = red[0]; __syncthreads();
    float sum = 0.0f;
    for (int t = tid; t < NS; t += 256) { float e = __expf(sc[t] - mx); sc[t] = e; sum += e; }
    red[tid] = sum; __syncthreads();
    for (int s = 128; s > 0; s >>= 1) {
        if (tid < s) red[tid] += red[tid + s];
        __syncthreads();
    }
    const float inv = 1.0f / red[0]; __syncthreads();

    // context[e] = sum_t attn[t] * enc_out[b,t,e]
    for (int e = tid; e < 512; e += 256) {
        float c = 0.0f;
        const float* eo = enc_out + ((size_t)b * NS) * 512 + e;
#pragma unroll 4
        for (int t = 0; t < NS; ++t) c += sc[t] * eo[(size_t)t * 512];
        c *= inv;
        ctx[b * 512 + e] = c;
        dec_in[b * KDEC + 1 + e] = f2bf(c);
    }
    if (tid == 0) dec_in[b * KDEC] = f2bf(y_buf[b]);
}

// ---------------------------------------------------------------------------
// Decoder LSTM cell + linear output: single WG, all 64 batch rows via WMMA.
//  z = dec_in @ Wd.T + zc ; h = sig(o)*tanh(sig(f)*c0 + sig(i)*tanh(g))
//  y[b] = [h, ctx] . W_lin + b_lin  -> out[b*512 + step], y_buf, h_attn
// ---------------------------------------------------------------------------
__global__ __launch_bounds__(1024) void dec_cell(
    const ushort_t* __restrict__ dec_in,  // [64][544] bf16
    const ushort_t* __restrict__ Wd,      // [1024][544] bf16
    const float* __restrict__ zc,         // [64][1024]
    const float* __restrict__ dec_c0,     // [64][256]
    const float* __restrict__ ctx,        // [64][512]
    const float* __restrict__ W_lin,      // [768]
    const float* __restrict__ b_lin,      // [1]
    float* __restrict__ h_attn,           // [64][256]
    float* __restrict__ y_buf,            // [64]
    float* __restrict__ out,              // [64][512]
    int step)
{
    __shared__ ushort_t h_sh[64][264];
    __shared__ float yred[64][16];

    const int tid = threadIdx.x;
    const int wave = tid >> 5, lane = tid & 31;
    const int mtile = wave >> 3, gpair = wave & 7;
    const int m0 = mtile * 16;
    const int lrow = lane & 15, lhalf = lane >> 4;
    const int col0[2] = { gpair * 32, gpair * 32 + 16 };

    // accumulators initialized with zc (= b_d + h0d @ Whh_d.T)
    v8f acc[2][4];
#pragma unroll
    for (int j = 0; j < 2; ++j)
#pragma unroll
        for (int g = 0; g < 4; ++g)
#pragma unroll
            for (int r = 0; r < 8; ++r) {
                int row = m0 + lhalf * 8 + r;
                acc[j][g][r] = zc[row * 1024 + g * 256 + col0[j] + lrow];
            }

    // unroll-1: keep fragment pressure bounded (acc already uses 64 VGPRs)
#pragma unroll 1
    for (int ks = 0; ks < 17; ++ks) {
        int kb = ks * 32 + lhalf * 16;
        const ushort_t* ap = &dec_in[(m0 + lrow) * KDEC];
        uint4 a0 = *(const uint4*)(ap + kb);
        uint4 a1 = *(const uint4*)(ap + kb + 8);
        v16bf afrag = make_frag(a0, a1);
#pragma unroll
        for (int j = 0; j < 2; ++j)
#pragma unroll
            for (int g = 0; g < 4; ++g) {
                const ushort_t* bp =
                    &Wd[(g * 256 + col0[j] + lrow) * KDEC + ks * 32 + lhalf * 16];
                uint4 b0 = *(const uint4*)(bp);
                uint4 b1 = *(const uint4*)(bp + 8);
                acc[j][g] = WMMA_BF16(afrag, make_frag(b0, b1), acc[j][g]);
            }
    }

#pragma unroll
    for (int j = 0; j < 2; ++j)
#pragma unroll
        for (int r = 0; r < 8; ++r) {
            int row = m0 + lhalf * 8 + r;
            int col = col0[j] + lrow;
            float iv = sigmoidf_(acc[j][0][r]);
            float fv = sigmoidf_(acc[j][1][r]);
            float gv = tanhf   (acc[j][2][r]);
            float ov = sigmoidf_(acc[j][3][r]);
            float c = fv * dec_c0[row * 256 + col] + iv * gv;
            float h = ov * tanhf(c);
            h_sh[row][col] = f2bf(h);
            h_attn[row * 256 + col] = h;
        }
    __syncthreads();

    // y[b] = h . W_lin[0:256] + ctx . W_lin[256:768] + b_lin, 16 threads per row
    {
        const int r = tid >> 4, part = tid & 15;
        float s = 0.0f;
        for (int k = part; k < 256; k += 16) s += bf2f(h_sh[r][k]) * W_lin[k];
        for (int k = part; k < 512; k += 16) s += ctx[r * 512 + k] * W_lin[256 + k];
        yred[r][part] = s;
    }
    __syncthreads();
    if ((tid & 15) == 0) {
        const int r = tid >> 4;
        float t = 0.0f;
#pragma unroll
        for (int i = 0; i < 16; ++i) t += yred[r][i];
        float y = t + b_lin[0];
        y_buf[r] = y;
        out[r * NS + step] = y;
    }
}

// ---------------------------------------------------------------------------
// Host launch
// ---------------------------------------------------------------------------
extern "C" void kernel_launch(void* const* d_in, const int* in_sizes, int n_in,
                              void* d_out, int out_size, void* d_ws, size_t ws_size,
                              hipStream_t stream) {
    (void)in_sizes; (void)n_in; (void)out_size; (void)ws_size;

    const float* x      = (const float*)d_in[0];
    const float* enc_h0 = (const float*)d_in[1];
    const float* enc_c0 = (const float*)d_in[2];
    const float* dec_h0 = (const float*)d_in[3];
    const float* dec_c0 = (const float*)d_in[4];
    const float* Wih_f  = (const float*)d_in[5];
    const float* Whh_f  = (const float*)d_in[6];
    const float* b_f    = (const float*)d_in[7];
    const float* Wih_b  = (const float*)d_in[8];
    const float* Whh_b  = (const float*)d_in[9];
    const float* b_b    = (const float*)d_in[10];
    const float* W_attn = (const float*)d_in[11];
    const float* b_attn = (const float*)d_in[12];
    const float* vvec   = (const float*)d_in[13];
    const float* Wih_d  = (const float*)d_in[14];
    const float* Whh_d  = (const float*)d_in[15];
    const float* b_d    = (const float*)d_in[16];
    const float* W_lin  = (const float*)d_in[17];
    const float* b_lin  = (const float*)d_in[18];
    float* out = (float*)d_out;

    // workspace carve-up (256B aligned)
    char* w = (char*)d_ws;
    auto carve = [&](size_t bytes) {
        void* p = (void*)w;
        w += (bytes + 255) & ~(size_t)255;
        return p;
    };
    ushort_t* x_bf     = (ushort_t*)carve((size_t)NB * NS * NI * 2);
    ushort_t* Wf       = (ushort_t*)carve((size_t)1024 * KENC * 2);
    ushort_t* Wb       = (ushort_t*)carve((size_t)1024 * KENC * 2);
    ushort_t* Wattn_e  = (ushort_t*)carve((size_t)256 * 512 * 2);
    ushort_t* Wd       = (ushort_t*)carve((size_t)1024 * KDEC * 2);
    float*    enc_out  = (float*)   carve((size_t)NB * NS * 512 * 4);
    ushort_t* enc_obf  = (ushort_t*)carve((size_t)NB * NS * 512 * 2);
    float*    enc_proj = (float*)   carve((size_t)NB * NS * 256 * 4);
    float*    zc       = (float*)   carve((size_t)NB * 1024 * 4);
    float*    h_attn   = (float*)   carve((size_t)NB * ND * 4);
    float*    y_buf    = (float*)   carve((size_t)NB * 4);
    float*    ctx      = (float*)   carve((size_t)NB * 512 * 4);
    ushort_t* dec_in   = (ushort_t*)carve((size_t)NB * KDEC * 2);

    // ---- prep ----
    k_cvt_x    <<<(NB * NS * NI + 255) / 256, 256, 0, stream>>>(x, x_bf);
    k_pack_w   <<<(1024 * KENC + 255) / 256, 256, 0, stream>>>(Wih_f, Whh_f, Wf);
    k_pack_w   <<<(1024 * KENC + 255) / 256, 256, 0, stream>>>(Wih_b, Whh_b, Wb);
    k_cvt_wattn<<<(256 * 512 + 255) / 256, 256, 0, stream>>>(W_attn, Wattn_e);
    k_cvt_wd   <<<(1024 * KDEC + 255) / 256, 256, 0, stream>>>(Wih_d, Wd);
    k_init_dec <<<(NB * KDEC + 255) / 256, 256, 0, stream>>>(dec_h0, h_attn, y_buf, dec_in);
    k_zc       <<<(NB * 1024 + 255) / 256, 256, 0, stream>>>(dec_h0, Whh_d, b_d, zc);

    // ---- encoder (both directions in parallel) ----
    enc_lstm<<<2, 1024, 0, stream>>>(x_bf, Wf, Wb, b_f, b_b, enc_h0, enc_c0,
                                     enc_out, enc_obf);

    // ---- attention projection (hoisted out of the decoder loop) ----
    proj_gemm<<<(NB * NS) / 8, 256, 0, stream>>>(enc_obf, Wattn_e, b_attn, enc_proj);

    // ---- decoder: 512 sequential steps ----
    for (int step = 0; step < NS; ++step) {
        attn_step<<<NB, 256, 0, stream>>>(enc_proj, enc_out, W_attn, vvec,
                                          h_attn, y_buf, ctx, dec_in);
        dec_cell<<<1, 1024, 0, stream>>>(dec_in, Wd, zc, dec_c0, ctx,
                                         W_lin, b_lin, h_attn, y_buf, out, step);
    }
}
